// GFormerWithSide_953482740023
// MI455X (gfx1250) — matrix operand order repaired and checked by reading, not compiled, not run on hardware
//
#include <hip/hip_runtime.h>
#include <cstdint>
#include <cstddef>

// ---------------- problem constants ----------------
#define NU_  30000
#define NI_  10000
#define ED_  64
#define H_   4
#define D_   256
#define NN_  40000           // NU_+NI_
#define EE_  100000
#define BB_  8192
#define HD_  1024            // H_*D_
#define NEGF (-1e30f)

// ---------------- CDNA5 WMMA types ----------------
typedef __attribute__((ext_vector_type(16))) __bf16        bf16x16;
typedef __attribute__((ext_vector_type(8)))  float         f32x8;
typedef __attribute__((ext_vector_type(8)))  unsigned int  u32x8;

// ---------------- small device helpers ----------------
__device__ __forceinline__ unsigned short f2bf(float f) {
  unsigned u = __float_as_uint(f);
  u += 0x7FFFu + ((u >> 16) & 1u);              // round-to-nearest-even
  return (unsigned short)(u >> 16);
}
__device__ __forceinline__ float bf2f(unsigned short h) {
  return __uint_as_float(((unsigned)h) << 16);
}
// monotonic float<->uint mapping for atomicMax-based segment max
__device__ __forceinline__ unsigned f2ord(float f) {
  unsigned u = __float_as_uint(f);
  return (u >> 31) ? ~u : (u | 0x80000000u);
}
__device__ __forceinline__ float ord2f(unsigned o) {
  unsigned u = (o >> 31) ? (o & 0x7FFFFFFFu) : ~o;
  return __uint_as_float(u);
}
__device__ __forceinline__ float sigmoidf(float x) { return 1.0f / (1.0f + expf(-x)); }

// ---------------- fills / converts ----------------
__global__ void fill_f32_k(float* p, float v, long n) {
  long i = (long)blockIdx.x * blockDim.x + threadIdx.x;
  if (i < n) p[i] = v;
}
__global__ void fill_ord_k(unsigned* p, float v, long n) {
  long i = (long)blockIdx.x * blockDim.x + threadIdx.x;
  if (i < n) p[i] = f2ord(v);
}
__global__ void cvt_f32_bf16_k(const float* __restrict__ in, unsigned short* __restrict__ out, long n) {
  long i = (long)blockIdx.x * blockDim.x + threadIdx.x;
  if (i < n) out[i] = f2bf(in[i]);
}
// concat [rows,256]|[rows,256] -> bf16 [rows,512]
__global__ void cat_bf16_k(const float* __restrict__ a, const float* __restrict__ b,
                           unsigned short* __restrict__ out, int rows) {
  long i = (long)blockIdx.x * blockDim.x + threadIdx.x;
  long total = (long)rows * 512;
  if (i >= total) return;
  int r = (int)(i >> 9);
  int c = (int)(i & 511);
  float v = (c < 256) ? a[(size_t)r * 256 + c] : b[(size_t)r * 256 + (c - 256)];
  out[i] = f2bf(v);
}

// ---------------- weight pre-pack into WMMA B-fragment layout ----------------
// B is f32 row-major [K,N]. Fragments are grouped so that the 8 n-tiles of one
// (128-col strip, kt) step are CONTIGUOUS (8 KB): frag id = (ns*KT + kt)*8 + j.
// Within a frag: lane L holds 8 dwords; n = ns*128 + j*16 + (L&15),
// khalf = L>>4, dword i holds K = kt*32 + 16*khalf + 2i (lo) / +1 (hi).
__global__ __launch_bounds__(256) void pack_weight_k(const float* __restrict__ B,
                                                     unsigned* __restrict__ out, int K, int N) {
  int KT = K >> 5;
  long total = (long)(N >> 4) * KT * 256;     // dwords
  long tid = (long)blockIdx.x * blockDim.x + threadIdx.x;
  if (tid >= total) return;
  int i    = (int)(tid & 7);
  int lane = (int)((tid >> 3) & 31);
  int j    = (int)((tid >> 8) & 7);           // n-tile within strip
  int kt   = (int)((tid >> 11) % KT);
  int ns   = (int)(tid / ((long)KT << 11));
  int n  = (ns << 7) + (j << 4) + (lane & 15);
  int k0 = (kt << 5) + ((lane >> 4) << 4) + (i << 1);
  unsigned lo = f2bf(B[(size_t)k0 * N + n]);
  unsigned hi = f2bf(B[(size_t)(k0 + 1) * N + n]);
  out[tid] = lo | (hi << 16);
}

// ---------------- WMMA GEMM: C[M,N] = A_bf16[M,K] @ Bpacked + bias ----------------
// Each wave owns a 16x128 strip (8 accumulators). Per K-step: one A fragment
// (2x b128), one clause of 8 contiguous B-fragment loads (immediate offsets
// j*1024 from a base that bumps by 8192/kt), then 8 back-to-back WMMAs.
__global__ __launch_bounds__(256) void wmma_gemm_k(
    const unsigned short* __restrict__ A, const u32x8* __restrict__ Bp,
    const float* __restrict__ bias, float* __restrict__ Cf,
    unsigned short* __restrict__ Cb, int M, int K, int N) {
  const int KT = K >> 5;
  const int MT = M >> 4;
  const int NS = N >> 7;                 // strips of 128 columns
  const int wave = blockIdx.x * (blockDim.x >> 5) + (threadIdx.x >> 5);
  if (wave >= MT * NS) return;
  const int lane = threadIdx.x & 31;
  const int mt = wave % MT;
  const int ns = wave / MT;
  const int half = lane >> 4;
  const int l16  = lane & 15;

  const f32x8 zero = {0.f,0.f,0.f,0.f,0.f,0.f,0.f,0.f};
  f32x8 acc[8];
#pragma unroll
  for (int j = 0; j < 8; ++j) acc[j] = zero;

  const unsigned short* Arow = A + (size_t)(mt * 16 + l16) * K;
  // strip base in u32x8 units: 8 frags * 32 lanes per kt step = 256
  const u32x8* bp = Bp + (size_t)ns * KT * 256 + lane;

  for (int kt = 0; kt < KT; ++kt) {
    const unsigned short* ap = Arow + (kt << 5) + (half << 4);
    if (kt + 1 < KT) __builtin_prefetch(ap + 32, 0, 3);   // global_prefetch_b8
    uint4 a0 = *(const uint4*)ap;
    uint4 a1 = *(const uint4*)(ap + 8);
    u32x8 au = {a0.x, a0.y, a0.z, a0.w, a1.x, a1.y, a1.z, a1.w};
    bf16x16 af = __builtin_bit_cast(bf16x16, au);
    const u32x8* bpk = bp + (size_t)kt * 256;
    u32x8 bu[8];
#pragma unroll
    for (int j = 0; j < 8; ++j)
      bu[j] = bpk[j * 32];                  // immediate offsets j*1024 bytes
#pragma unroll
    for (int j = 0; j < 8; ++j)
      acc[j] = __builtin_amdgcn_wmma_f32_16x16x32_bf16(
          false, af, false, __builtin_bit_cast(bf16x16, bu[j]), (short)0,
          acc[j], false, false);
  }

  const int row0 = (mt << 4) + (half << 3);
  const int col0 = (ns << 7) + l16;
#pragma unroll
  for (int j = 0; j < 8; ++j) {
    int col = col0 + (j << 4);
    float bv = bias ? bias[col] : 0.0f;
#pragma unroll
    for (int r = 0; r < 8; ++r) {
      float v = acc[j][r] + bv;
      size_t idx = (size_t)(row0 + r) * N + col;
      if (Cb) Cb[idx] = f2bf(v);
      else    Cf[idx] = v;
    }
  }
}

// ---------------- wine side-feature attention (4 tokens) ----------------
__global__ __launch_bounds__(256) void side_attn_k(const float* __restrict__ q,
                                                   const float* __restrict__ k,
                                                   const float* __restrict__ v,
                                                   float* __restrict__ ao) {
  int n = blockIdx.x;
  int t = threadIdx.x;
  __shared__ float red[256];
  __shared__ float at[4];
  float qv = q[(size_t)n * 256 + t];
#pragma unroll
  for (int s = 0; s < 4; ++s) {
    red[t] = qv * k[((size_t)n * 4 + s) * 256 + t];
    __syncthreads();
    for (int o = 128; o; o >>= 1) {
      if (t < o) red[t] += red[t + o];
      __syncthreads();
    }
    if (t == 0) at[s] = red[0] * (1.0f / 16.0f);   // / sqrt(256)
    __syncthreads();
  }
  if (t == 0) {
    float mx = fmaxf(fmaxf(at[0], at[1]), fmaxf(at[2], at[3]));
    float e0 = expf(at[0] - mx), e1 = expf(at[1] - mx);
    float e2 = expf(at[2] - mx), e3 = expf(at[3] - mx);
    float s = e0 + e1 + e2 + e3;
    at[0] = e0 / s; at[1] = e1 / s; at[2] = e2 / s; at[3] = e3 / s;
  }
  __syncthreads();
  float o = 0.f;
#pragma unroll
  for (int s = 0; s < 4; ++s) o += at[s] * v[((size_t)n * 4 + s) * 256 + t];
  ao[(size_t)n * 256 + t] = o;
}

// ---------------- GCN ----------------
__global__ void edge_deg_k(const int* __restrict__ dst, const unsigned char* __restrict__ keep,
                           float* __restrict__ deg) {
  int e = blockIdx.x * blockDim.x + threadIdx.x;
  if (e < EE_ && keep[e]) atomicAdd(&deg[dst[e]], 1.0f);
}
__global__ void dinv_k(const float* __restrict__ deg, float* __restrict__ dinv) {
  int i = blockIdx.x * blockDim.x + threadIdx.x;
  if (i < NN_) dinv[i] = rsqrtf(deg[i]);
}
__global__ void x1_init_k(const float* __restrict__ xw, const float* __restrict__ dinv,
                          const float* __restrict__ g_b, float* __restrict__ x1) {
  long idx = (long)blockIdx.x * blockDim.x + threadIdx.x;
  if (idx >= (long)NN_ * D_) return;
  int n = (int)(idx >> 8);
  int d = (int)(idx & 255);
  float di = dinv[n];
  x1[idx] = xw[idx] * di * di + g_b[d];
}
__global__ __launch_bounds__(256) void gcn_scatter_k(const int* __restrict__ src,
                                                     const int* __restrict__ dst,
                                                     const unsigned char* __restrict__ keep,
                                                     const float* __restrict__ dinv,
                                                     const float* __restrict__ xw,
                                                     float* __restrict__ x1) {
  int e = blockIdx.x;
  if (!keep[e]) return;
  int s = src[e], dd = dst[e];
  float nrm = dinv[s] * dinv[dd];
  int t = threadIdx.x;
  atomicAdd(&x1[(size_t)dd * 256 + t], xw[(size_t)s * 256 + t] * nrm);
}

// ---------------- TransformerConv edge attention ----------------
__global__ __launch_bounds__(256) void edge_logits_k(const int* __restrict__ src,
                                                     const int* __restrict__ dst,
                                                     const unsigned char* __restrict__ keep,
                                                     const unsigned short* __restrict__ qh,
                                                     const unsigned short* __restrict__ kh,
                                                     float* __restrict__ logits,
                                                     unsigned* __restrict__ mbuf) {
  int gw = (blockIdx.x * blockDim.x + threadIdx.x) >> 5;
  int lane = threadIdx.x & 31;
  if (gw >= EE_ * H_) return;
  int e = gw >> 2, h = gw & 3;
  int s = src[e], dd = dst[e];
  const unsigned short* qp = qh + (size_t)dd * HD_ + h * D_;
  const unsigned short* kp = kh + (size_t)s * HD_ + h * D_;
  float acc = 0.f;
  for (int i = lane; i < D_; i += 32) acc += bf2f(qp[i]) * bf2f(kp[i]);
  for (int o = 16; o; o >>= 1) acc += __shfl_down(acc, o, 32);
  if (lane == 0) {
    float lg = keep[e] ? acc * (1.0f / 16.0f) : NEGF;
    logits[e * H_ + h] = lg;
    atomicMax(mbuf + (size_t)dd * H_ + h, f2ord(lg));
  }
}
__global__ void m_final_k(const unsigned* __restrict__ mbuf, float* __restrict__ mfin) {
  int i = blockIdx.x * blockDim.x + threadIdx.x;
  if (i >= NN_ * H_) return;
  float m = ord2f(mbuf[i]);
  mfin[i] = (m < NEGF * 0.5f) ? 0.0f : m;
}
__global__ __launch_bounds__(256) void edge_scatter_k(const int* __restrict__ src,
                                                      const int* __restrict__ dst,
                                                      const unsigned char* __restrict__ keep,
                                                      const float* __restrict__ logits,
                                                      const float* __restrict__ mfin,
                                                      const unsigned short* __restrict__ vh,
                                                      float* __restrict__ num,
                                                      float* __restrict__ den) {
  int e = blockIdx.x;
  if (!keep[e]) return;
  int s = src[e], dd = dst[e];
  int t = threadIdx.x;
#pragma unroll
  for (int h = 0; h < H_; ++h) {
    float eh = expf(logits[e * H_ + h] - mfin[(size_t)dd * H_ + h]);
    float vv = bf2f(vh[(size_t)s * HD_ + h * D_ + t]);
    atomicAdd(&num[((size_t)dd * H_ + h) * D_ + t], eh * vv);
    if (t == 0) atomicAdd(&den[(size_t)dd * H_ + h], eh);
  }
}
__global__ void x2_final_k(const float* __restrict__ num, const float* __restrict__ den,
                           const float* __restrict__ skip, float* __restrict__ x2) {
  long idx = (long)blockIdx.x * blockDim.x + threadIdx.x;
  if (idx >= (long)NN_ * D_) return;
  int n = (int)(idx >> 8);
  int d = (int)(idx & 255);
  float acc = 0.f;
#pragma unroll
  for (int h = 0; h < H_; ++h)
    acc += num[((size_t)n * H_ + h) * D_ + d] / fmaxf(den[(size_t)n * H_ + h], 1e-16f);
  x2[idx] = 0.25f * acc + skip[idx];
}

// ---------------- losses + scores ----------------
__global__ __launch_bounds__(256) void recon_partial_k(const float* __restrict__ x2,
                                                       const unsigned char* __restrict__ mask,
                                                       const float* __restrict__ ae_b,
                                                       float* __restrict__ accum) {
  __shared__ float sh[256];
  long idx = (long)blockIdx.x * blockDim.x + threadIdx.x;
  float v = 0.f;
  if (idx < (long)NN_ * D_) {
    int n = (int)(idx >> 8);
    int d = (int)(idx & 255);
    if (mask[n]) { float df = ae_b[d] - x2[idx]; v = df * df; }
  }
  int t = threadIdx.x;
  sh[t] = v;
  __syncthreads();
  for (int o = 128; o; o >>= 1) { if (t < o) sh[t] += sh[t + o]; __syncthreads(); }
  if (t == 0) atomicAdd(&accum[0], sh[0]);
}
__global__ void mask_count_k(const unsigned char* __restrict__ mask, float* __restrict__ accum) {
  int i = blockIdx.x * blockDim.x + threadIdx.x;
  if (i < NN_ && mask[i]) atomicAdd(&accum[1], 1.0f);
}
__global__ __launch_bounds__(256) void score_k(const int* __restrict__ user,
                                               const int* __restrict__ wine,
                                               const float* __restrict__ x2,
                                               const float* __restrict__ ep_w,
                                               const float* __restrict__ ep_b,
                                               const float* __restrict__ fc_w,
                                               const float* __restrict__ fc_b,
                                               float* __restrict__ out,
                                               float* __restrict__ accum) {
  __shared__ float s1[256];
  __shared__ float s2[256];
  int b = blockIdx.x;
  int t = threadIdx.x;
  const float* u = x2 + (size_t)user[b] * D_;
  const float* w = x2 + (size_t)(wine[b] + NU_) * D_;
  float uv = u[t], wv = w[t];
  s1[t] = uv * ep_w[t] + wv * ep_w[256 + t];
  s2[t] = uv * fc_w[t] + wv * fc_w[256 + t];
  __syncthreads();
  for (int o = 128; o; o >>= 1) {
    if (t < o) { s1[t] += s1[t + o]; s2[t] += s2[t + o]; }
    __syncthreads();
  }
  if (t == 0) {
    float es = sigmoidf(s1[0] + ep_b[0]);
    float d = es - 1.0f;
    atomicAdd(&accum[2], d * d);
    out[b] = sigmoidf(s2[0] + fc_b[0]) * 4.0f + 1.0f;
  }
}
__global__ void finalize_losses_k(const float* __restrict__ accum, float* __restrict__ out) {
  out[BB_]     = accum[0] / (fmaxf(accum[1], 1.0f) * (float)D_);
  out[BB_ + 1] = accum[2] / (float)BB_;
}

// ---------------- host launcher ----------------
static inline int div_up_l(long a, int b) { return (int)((a + b - 1) / b); }

extern "C" void kernel_launch(void* const* d_in, const int* in_sizes, int n_in,
                              void* d_out, int out_size, void* d_ws, size_t ws_size,
                              hipStream_t stream) {
  (void)in_sizes; (void)n_in; (void)out_size;
  const int* user = (const int*)d_in[0];
  const int* wine = (const int*)d_in[1];
  const int* ei   = (const int*)d_in[2];
  const int* src  = ei;
  const int* dst  = ei + EE_;
  const unsigned char* keep  = (const unsigned char*)d_in[3];
  const unsigned char* nmask = (const unsigned char*)d_in[4];
  const float* user_emb = (const float*)d_in[5];
  const float* wine_emb = (const float*)d_in[6];
  const float* up_w = (const float*)d_in[7];   const float* up_b = (const float*)d_in[8];
  const float* wp_w = (const float*)d_in[9];   const float* wp_b = (const float*)d_in[10];
  const float* sf_proj_w = (const float*)d_in[11]; const float* sf_proj_b = (const float*)d_in[12];
  const float* sf_q_w = (const float*)d_in[13]; const float* sf_q_b = (const float*)d_in[14];
  const float* sf_k_w = (const float*)d_in[15]; const float* sf_k_b = (const float*)d_in[16];
  const float* sf_v_w = (const float*)d_in[17]; const float* sf_v_b = (const float*)d_in[18];
  const float* sf_o_w = (const float*)d_in[19]; const float* sf_o_b = (const float*)d_in[20];
  const float* sf_fc_w = (const float*)d_in[21]; const float* sf_fc_b = (const float*)d_in[22];
  const float* g_w = (const float*)d_in[23];   const float* g_b = (const float*)d_in[24];
  const float* t_q_w = (const float*)d_in[25]; const float* t_q_b = (const float*)d_in[26];
  const float* t_k_w = (const float*)d_in[27]; const float* t_k_b = (const float*)d_in[28];
  const float* t_v_w = (const float*)d_in[29]; const float* t_v_b = (const float*)d_in[30];
  const float* t_skip_w = (const float*)d_in[31]; const float* t_skip_b = (const float*)d_in[32];
  const float* fc_w = (const float*)d_in[33];  const float* fc_b = (const float*)d_in[34];
  /* ae_w unused (recon uses bias only) */     const float* ae_b = (const float*)d_in[36];
  const float* ep_w = (const float*)d_in[37];  const float* ep_b = (const float*)d_in[38];
  const float* wsf  = (const float*)d_in[39];
  float* out = (float*)d_out;

  // ---- workspace bump allocator ----
  char* wsp = (char*)d_ws;
  size_t off = 0;
  auto alloc = [&](size_t bytes) -> void* {
    void* p = wsp + off;
    off = (off + bytes + 255) & ~(size_t)255;
    return p;
  };
  // bf16 activations
  unsigned short* ue_bf    = (unsigned short*)alloc((size_t)NU_ * ED_ * 2);
  unsigned short* we_bf    = (unsigned short*)alloc((size_t)NI_ * ED_ * 2);
  unsigned short* wsf_bf   = (unsigned short*)alloc((size_t)NN_ * ED_ * 2);   // [NI*4, 64]
  unsigned short* wproj_bf = (unsigned short*)alloc((size_t)NI_ * D_ * 2);
  unsigned short* side_bf  = (unsigned short*)alloc((size_t)NN_ * D_ * 2);
  unsigned short* aopre_bf = (unsigned short*)alloc((size_t)NI_ * D_ * 2);
  unsigned short* cat_bf   = (unsigned short*)alloc((size_t)NI_ * 2 * D_ * 2);
  unsigned short* x0_bf    = (unsigned short*)alloc((size_t)NN_ * D_ * 2);
  unsigned short* x1_bf    = (unsigned short*)alloc((size_t)NN_ * D_ * 2);
  // packed weights (bf16 fragment layout)
  unsigned* p_up    = (unsigned*)alloc((size_t)ED_ * D_ * 2);
  unsigned* p_wp    = (unsigned*)alloc((size_t)ED_ * D_ * 2);
  unsigned* p_sfpr  = (unsigned*)alloc((size_t)ED_ * D_ * 2);
  unsigned* p_sfq   = (unsigned*)alloc((size_t)D_ * D_ * 2);
  unsigned* p_sfk   = (unsigned*)alloc((size_t)D_ * D_ * 2);
  unsigned* p_sfv   = (unsigned*)alloc((size_t)D_ * D_ * 2);
  unsigned* p_sfo   = (unsigned*)alloc((size_t)D_ * D_ * 2);
  unsigned* p_sffc  = (unsigned*)alloc((size_t)2 * D_ * D_ * 2);
  unsigned* p_g     = (unsigned*)alloc((size_t)D_ * D_ * 2);
  unsigned* p_tq    = (unsigned*)alloc((size_t)D_ * HD_ * 2);
  unsigned* p_tk    = (unsigned*)alloc((size_t)D_ * HD_ * 2);
  unsigned* p_tv    = (unsigned*)alloc((size_t)D_ * HD_ * 2);
  unsigned* p_tsk   = (unsigned*)alloc((size_t)D_ * D_ * 2);
  // f32 activations
  float* x0    = (float*)alloc((size_t)NN_ * D_ * 4);
  float* wproj = (float*)alloc((size_t)NI_ * D_ * 4);
  float* side  = (float*)alloc((size_t)NN_ * D_ * 4);
  float* qbuf  = (float*)alloc((size_t)NI_ * D_ * 4);
  float* kbuf  = (float*)alloc((size_t)NN_ * D_ * 4);
  float* vbuf  = (float*)alloc((size_t)NN_ * D_ * 4);
  float* aopre = (float*)alloc((size_t)NI_ * D_ * 4);
  float* aobuf = (float*)alloc((size_t)NI_ * D_ * 4);
  float* deg   = (float*)alloc((size_t)NN_ * 4);
  float* dinv  = (float*)alloc((size_t)NN_ * 4);
  float* xw    = (float*)alloc((size_t)NN_ * D_ * 4);
  float* x1    = (float*)alloc((size_t)NN_ * D_ * 4);
  unsigned short* qh_bf = (unsigned short*)alloc((size_t)NN_ * HD_ * 2);
  unsigned short* kh_bf = (unsigned short*)alloc((size_t)NN_ * HD_ * 2);
  unsigned short* vh_bf = (unsigned short*)alloc((size_t)NN_ * HD_ * 2);
  float* skip   = (float*)alloc((size_t)NN_ * D_ * 4);
  float* logits = (float*)alloc((size_t)EE_ * H_ * 4);
  unsigned* mbuf = (unsigned*)alloc((size_t)NN_ * H_ * 4);
  float* mfin   = (float*)alloc((size_t)NN_ * H_ * 4);
  float* den    = (float*)alloc((size_t)NN_ * H_ * 4);
  float* num    = (float*)alloc((size_t)NN_ * H_ * D_ * 4);
  float* x2     = (float*)alloc((size_t)NN_ * D_ * 4);
  float* accum  = (float*)alloc(256);
  if (off > ws_size) return;   // insufficient scratch: deterministic no-op

  auto fillf = [&](float* p, float v, long n) {
    fill_f32_k<<<div_up_l(n, 256), 256, 0, stream>>>(p, v, n);
  };
  auto cvt = [&](const float* in, unsigned short* o, long n) {
    cvt_f32_bf16_k<<<div_up_l(n, 256), 256, 0, stream>>>(in, o, n);
  };
  auto pack = [&](const float* w, unsigned* o, int K, int N) {
    long tot = (long)(N >> 4) * (K >> 5) * 256;
    pack_weight_k<<<div_up_l(tot, 256), 256, 0, stream>>>(w, o, K, N);
  };
  auto gemm = [&](const unsigned short* A, const unsigned* Bp, const float* bias,
                  float* Cf, unsigned short* Cb, int M, int K, int N) {
    int waves = (M >> 4) * (N >> 7);      // 16x128 strip per wave
    wmma_gemm_k<<<(waves + 7) / 8, 256, 0, stream>>>(
        A, (const u32x8*)Bp, bias, Cf, Cb, M, K, N);
  };

  // ---- init accumulators / scatter targets ----
  fillf(accum, 0.0f, 3);
  fillf(deg, 1.0f, NN_);
  fillf(den, 0.0f, (long)NN_ * H_);
  fillf(num, 0.0f, (long)NN_ * H_ * D_);
  fill_ord_k<<<div_up_l((long)NN_ * H_, 256), 256, 0, stream>>>(mbuf, NEGF, (long)NN_ * H_);

  // ---- bf16 conversions + weight packing ----
  cvt(user_emb, ue_bf, (long)NU_ * ED_);
  cvt(wine_emb, we_bf, (long)NI_ * ED_);
  cvt(wsf, wsf_bf, (long)NN_ * ED_);
  pack(up_w, p_up, ED_, D_);      pack(wp_w, p_wp, ED_, D_);
  pack(sf_proj_w, p_sfpr, ED_, D_);
  pack(sf_q_w, p_sfq, D_, D_);    pack(sf_k_w, p_sfk, D_, D_);
  pack(sf_v_w, p_sfv, D_, D_);    pack(sf_o_w, p_sfo, D_, D_);
  pack(sf_fc_w, p_sffc, 2 * D_, D_);
  pack(g_w, p_g, D_, D_);
  pack(t_q_w, p_tq, D_, HD_);     pack(t_k_w, p_tk, D_, HD_);
  pack(t_v_w, p_tv, D_, HD_);     pack(t_skip_w, p_tsk, D_, D_);

  // ---- projections ----
  gemm(ue_bf, p_up, up_b, x0, nullptr, NU_, ED_, D_);          // u_proj -> x0 rows [0,NU)
  gemm(we_bf, p_wp, wp_b, wproj, nullptr, NI_, ED_, D_);       // w_proj
  gemm(wsf_bf, p_sfpr, sf_proj_b, side, nullptr, NN_, ED_, D_);// side tokens
  cvt(wproj, wproj_bf, (long)NI_ * D_);
  cvt(side, side_bf, (long)NN_ * D_);

  // ---- side fusion attention ----
  gemm(wproj_bf, p_sfq, sf_q_b, qbuf, nullptr, NI_, D_, D_);
  gemm(side_bf,  p_sfk, sf_k_b, kbuf, nullptr, NN_, D_, D_);
  gemm(side_bf,  p_sfv, sf_v_b, vbuf, nullptr, NN_, D_, D_);
  side_attn_k<<<NI_, 256, 0, stream>>>(qbuf, kbuf, vbuf, aopre);
  cvt(aopre, aopre_bf, (long)NI_ * D_);
  gemm(aopre_bf, p_sfo, sf_o_b, aobuf, nullptr, NI_, D_, D_);
  cat_bf16_k<<<div_up_l((long)NI_ * 512, 256), 256, 0, stream>>>(wproj, aobuf, cat_bf, NI_);
  gemm(cat_bf, p_sffc, sf_fc_b, x0 + (size_t)NU_ * D_, nullptr, NI_, 2 * D_, D_); // fused -> x0 rows [NU,N)

  // ---- GCNConv ----
  cvt(x0, x0_bf, (long)NN_ * D_);
  edge_deg_k<<<div_up_l(EE_, 256), 256, 0, stream>>>(dst, keep, deg);
  dinv_k<<<div_up_l(NN_, 256), 256, 0, stream>>>(deg, dinv);
  gemm(x0_bf, p_g, nullptr, xw, nullptr, NN_, D_, D_);
  x1_init_k<<<div_up_l((long)NN_ * D_, 256), 256, 0, stream>>>(xw, dinv, g_b, x1);
  gcn_scatter_k<<<EE_, 256, 0, stream>>>(src, dst, keep, dinv, xw, x1);

  // ---- TransformerConv ----
  cvt(x1, x1_bf, (long)NN_ * D_);
  gemm(x1_bf, p_tq, t_q_b, nullptr, qh_bf, NN_, D_, HD_);
  gemm(x1_bf, p_tk, t_k_b, nullptr, kh_bf, NN_, D_, HD_);
  gemm(x1_bf, p_tv, t_v_b, nullptr, vh_bf, NN_, D_, HD_);
  gemm(x1_bf, p_tsk, t_skip_b, skip, nullptr, NN_, D_, D_);
  edge_logits_k<<<div_up_l((long)EE_ * H_ * 32, 256), 256, 0, stream>>>(
      src, dst, keep, qh_bf, kh_bf, logits, mbuf);
  m_final_k<<<div_up_l((long)NN_ * H_, 256), 256, 0, stream>>>(mbuf, mfin);
  edge_scatter_k<<<EE_, 256, 0, stream>>>(src, dst, keep, logits, mfin, vh_bf, num, den);
  x2_final_k<<<div_up_l((long)NN_ * D_, 256), 256, 0, stream>>>(num, den, skip, x2);

  // ---- losses + scores ----
  recon_partial_k<<<div_up_l((long)NN_ * D_, 256), 256, 0, stream>>>(x2, nmask, ae_b, accum);
  mask_count_k<<<div_up_l(NN_, 256), 256, 0, stream>>>(nmask, accum);
  score_k<<<BB_, 256, 0, stream>>>(user, wine, x2, ep_w, ep_b, fc_w, fc_b, out, accum);
  finalize_losses_k<<<1, 1, 0, stream>>>(accum, out);
}
